// MultiHeadAttention_79465484911033
// MI455X (gfx1250) — matrix operand order repaired and checked by reading, not compiled, not run on hardware
//
#include <hip/hip_runtime.h>
#include <hip/hip_bf16.h>
#include <cstdint>

typedef __bf16 bf16;
typedef __attribute__((ext_vector_type(16))) bf16  v16bf;
typedef __attribute__((ext_vector_type(8)))  bf16  v8bf;
typedef __attribute__((ext_vector_type(8)))  float v8f;
typedef __attribute__((ext_vector_type(4)))  int   v4i;

#define BATCH   2
#define SEQ     2048
#define HDIM    1024
#define NHEADS  16
#define HD      64
#define QKV_N   3072
#define LOG1E4  9.210340371976184f

// ---------------------------------------------------------------------------
// CDNA5 async global->LDS copy (ASYNCcnt-tracked), with guarded fallback
// ---------------------------------------------------------------------------
#if __has_builtin(__builtin_amdgcn_global_load_async_to_lds_b128)
#define HAS_ASYNC_LDS 1
#else
#define HAS_ASYNC_LDS 0
#endif

#if __has_builtin(__builtin_amdgcn_s_wait_asynccnt)
#define WAIT_ASYNCCNT(n) __builtin_amdgcn_s_wait_asynccnt(n)
#else
#define WAIT_ASYNCCNT(n) asm volatile("s_wait_asynccnt " #n ::: "memory")
#endif

// builtin signature (from hipcc diagnostic): (v4i addrspace(1)*, v4i addrspace(3)*, imm, imm)
typedef __attribute__((address_space(1))) v4i* gv4p;
typedef __attribute__((address_space(3))) v4i* lv4p;

// copy 32 bytes (16 bf16) global -> LDS
__device__ __forceinline__ void copy32_async(const bf16* g, bf16* l) {
#if HAS_ASYNC_LDS
  __builtin_amdgcn_global_load_async_to_lds_b128(
      (gv4p)(__UINTPTR_TYPE__)(g),     (lv4p)(__UINTPTR_TYPE__)(l),     0, 0);
  __builtin_amdgcn_global_load_async_to_lds_b128(
      (gv4p)(__UINTPTR_TYPE__)(g + 8), (lv4p)(__UINTPTR_TYPE__)(l + 8), 0, 0);
#else
  *(v8bf*)l       = *(const v8bf*)g;
  *((v8bf*)l + 1) = *((const v8bf*)g + 1);
#endif
}

// ---------------------------------------------------------------------------
// f32 -> bf16 conversion
// ---------------------------------------------------------------------------
__global__ void cvt_bf16_kernel(const float* __restrict__ in,
                                bf16* __restrict__ out, int n) {
  int i = blockIdx.x * blockDim.x + threadIdx.x;
  if (i < n) out[i] = (bf16)in[i];
}

// ---------------------------------------------------------------------------
// WMMA helpers
// ---------------------------------------------------------------------------
__device__ __forceinline__ v16bf load_a_frag(const bf16* rowptr, int koff) {
  // A-layout (16x32 bf16): lane l -> row l&15; half=l>>4; elems 0-15 = K 2*koff..2*koff+15,
  // koff = 8*half (contiguous, matches B fragments).
  v8bf lo = *(const v8bf*)(rowptr + 2 * koff);
  v8bf hi = *(const v8bf*)(rowptr + 2 * koff + 8);
  return __builtin_shufflevector(lo, hi, 0,1,2,3,4,5,6,7,8,9,10,11,12,13,14,15);
}

__device__ __forceinline__ float hmax16(float v) {
#pragma unroll
  for (int m = 1; m < 16; m <<= 1) v = fmaxf(v, __shfl_xor(v, m, 32));
  return v;
}
__device__ __forceinline__ float hsum16(float v) {
#pragma unroll
  for (int m = 1; m < 16; m <<= 1) v += __shfl_xor(v, m, 32);
  return v;
}

// ---------------------------------------------------------------------------
// C[M,N] f32 = A[M,K]bf16 @ W[N,K]bf16^T + bias
// Block tile 128x128, K-step 32, double-buffered LDS fed by async copies.
// 8 waves: wave (wy in 0..3, wx in 0..1) computes 32 rows x 64 cols.
// ---------------------------------------------------------------------------
__global__ __launch_bounds__(256)
void gemm_lds_kernel(const bf16* __restrict__ A, const bf16* __restrict__ W,
                     const float* __restrict__ bias, float* __restrict__ C,
                     int M, int N, int K) {
  __shared__ __align__(32) bf16 ldsA[2][128 * 32];
  __shared__ __align__(32) bf16 ldsB[2][128 * 32];

  int tilesN = N >> 7;
  int tm = blockIdx.x / tilesN;
  int tn = blockIdx.x - tm * tilesN;
  int row0 = tm << 7, col0 = tn << 7;

  int tid = threadIdx.x;
  int wid = tid >> 5, lane = tid & 31;
  int lr = lane & 15, hf = lane >> 4;
  int wy = wid >> 1, wx = wid & 1;

  // cooperative copy mapping: thread -> (row r, 16-half chunk)
  int cr = tid >> 1;             // 0..127
  int cb = (tid & 1) * 16;       // 0 or 16 halves
  const bf16* gA = A + (size_t)(row0 + cr) * K + cb;
  const bf16* gW = W + (size_t)(col0 + cr) * K + cb;
  int ldso = cr * 32 + cb;

  v8f acc[2][4] = {};

  // prologue: stage K-tile 0
  copy32_async(gA, &ldsA[0][ldso]);
  copy32_async(gW, &ldsB[0][ldso]);

  const int NK = K >> 5;
  for (int kt = 0; kt < NK; ++kt) {
    int cur = kt & 1;
    if (kt + 1 < NK) {
      int ko = (kt + 1) << 5;
      copy32_async(gA + ko, &ldsA[cur ^ 1][ldso]);
      copy32_async(gW + ko, &ldsB[cur ^ 1][ldso]);
      WAIT_ASYNCCNT(4);          // tile kt's 4 copies done; next 4 in flight
    } else {
      WAIT_ASYNCCNT(0);
    }
    __syncthreads();

    v16bf af0 = load_a_frag(&ldsA[cur][(wy * 32 + lr) * 32], hf * 8);
    v16bf af1 = load_a_frag(&ldsA[cur][(wy * 32 + 16 + lr) * 32], hf * 8);
#pragma unroll
    for (int c = 0; c < 4; ++c) {
      v16bf bfr = *(const v16bf*)&ldsB[cur][(wx * 64 + c * 16 + lr) * 32 + hf * 16];
      acc[0][c] = __builtin_amdgcn_wmma_f32_16x16x32_bf16(false, af0, false, bfr, (short)0, acc[0][c], false, false);
      acc[1][c] = __builtin_amdgcn_wmma_f32_16x16x32_bf16(false, af1, false, bfr, (short)0, acc[1][c], false, false);
    }
    __syncthreads();             // protect buffer 'cur' before tile kt+2 overwrites it
  }

  // epilogue: C-layout -> lane holds col lr, VGPR v holds row v + 8*hf
#pragma unroll
  for (int mt = 0; mt < 2; ++mt) {
#pragma unroll
    for (int c = 0; c < 4; ++c) {
      int col = col0 + wx * 64 + c * 16 + lr;
      float bv = bias[col];
#pragma unroll
      for (int v = 0; v < 8; ++v) {
        int row = row0 + wy * 32 + mt * 16 + v + 8 * hf;
        C[(size_t)row * N + col] = acc[mt][c][v] + bv;
      }
    }
  }
}

// ---------------------------------------------------------------------------
// RoPE (2D spatial + 1D temporal composed into a single rotation per pair)
// qkv f32 (B,L,16,192) -> q,k bf16 (B,n,L,64), v bf16 transposed (B,n,64,L)
// ---------------------------------------------------------------------------
__global__ void rope_kernel(const float* __restrict__ qkv,
                            bf16* __restrict__ qo, bf16* __restrict__ ko,
                            bf16* __restrict__ vT) {
  int idx = blockIdx.x * blockDim.x + threadIdx.x;
  if (idx >= BATCH * SEQ * NHEADS * 32) return;
  int j = idx & 31;            // pair index (32 pairs of hd=64)
  int h = (idx >> 5) & 15;     // head
  int l = (idx >> 9) & (SEQ - 1);
  int b = idx >> 20;

  int ti = l >> 8;             // temporal position (t=8)
  int si = l & 255;            // spatial position (16x16)
  int sx = si & 15, sy = si >> 4;

  float ang_s = (j < 16) ? (float)sx * __expf(-(float)j * (LOG1E4 / 16.f))
                         : (float)sy * __expf(-(float)(j - 16) * (LOG1E4 / 16.f));
  float ang = ang_s + (float)ti * __expf(-(float)j * (LOG1E4 / 32.f));
  float cs = __cosf(ang), sn = __sinf(ang);

  size_t base = ((size_t)(b * SEQ + l)) * QKV_N + h * 192 + 2 * j;
  float qr = qkv[base],       qi = qkv[base + 1];
  float kr = qkv[base + 64],  ki = qkv[base + 65];
  float v0 = qkv[base + 128], v1 = qkv[base + 129];

  size_t bh = (size_t)b * NHEADS + h;
  size_t qidx = (bh * SEQ + l) * HD + 2 * j;
  qo[qidx]     = (bf16)(qr * cs - qi * sn);
  qo[qidx + 1] = (bf16)(qr * sn + qi * cs);
  ko[qidx]     = (bf16)(kr * cs - ki * sn);
  ko[qidx + 1] = (bf16)(kr * sn + ki * cs);
  vT[(bh * HD + 2 * j) * SEQ + l]     = (bf16)v0;
  vT[(bh * HD + 2 * j + 1) * SEQ + l] = (bf16)v1;
}

// ---------------------------------------------------------------------------
// Flash attention, causal. grid = B*n*(L/128); 8 waves/block; 1 wave = 16 q-rows.
// Epilogue applies SiLU and writes o_act bf16 in (B,L,H) layout.
// ---------------------------------------------------------------------------
__global__ __launch_bounds__(256)
void attn_kernel(const bf16* __restrict__ q, const bf16* __restrict__ k,
                 const bf16* __restrict__ vT, bf16* __restrict__ o_act) {
  __shared__ __align__(16) bf16 lds_p[8][16 * 32];

  int wid = threadIdx.x >> 5, lane = threadIdx.x & 31;
  int blk = blockIdx.x;
  int qt = blk & 15;
  int h  = (blk >> 4) & 15;
  int b  = blk >> 8;
  int qbase = qt * 128 + wid * 16;

  size_t bh = (size_t)b * NHEADS + h;
  const bf16* qp = q  + bh * SEQ * HD;
  const bf16* kp = k  + bh * SEQ * HD;
  const bf16* vp = vT + bh * HD * SEQ;

  int lr = lane & 15, hf = lane >> 4;
  int koff = hf * 8;

  const bf16* qrow = qp + (size_t)(qbase + lr) * HD;
  v16bf a0 = load_a_frag(qrow, koff);        // hd dims 0..31
  v16bf a1 = load_a_frag(qrow + 32, koff);   // hd dims 32..63

  float mrow[8], lrow[8];
#pragma unroll
  for (int v = 0; v < 8; ++v) { mrow[v] = -1e30f; lrow[v] = 0.f; }
  v8f o0 = {}, o1 = {}, o2 = {}, o3 = {};
  bf16* pl = lds_p[wid];
  const float scl = 0.125f;                  // 1/sqrt(64)

  for (int kb0 = 0; kb0 <= qbase + 15; kb0 += 32) {
    // prefetch next key block
    __builtin_prefetch(kp + (size_t)(kb0 + 32 + lr) * HD, 0, 0);
    // S = Q K^T for keys kb0..kb0+31 (two 16-key tiles)
    const bf16* kr0 = kp + (size_t)(kb0 + lr) * HD + hf * 16;
    const bf16* kr1 = kr0 + 16 * HD;
    v16bf b0lo = *(const v16bf*)(kr0);       // kdims 0..31
    v16bf b0hi = *(const v16bf*)(kr0 + 32);  // kdims 32..63
    v16bf b1lo = *(const v16bf*)(kr1);
    v16bf b1hi = *(const v16bf*)(kr1 + 32);
    v8f s0 = {}, s1 = {};
    s0 = __builtin_amdgcn_wmma_f32_16x16x32_bf16(false, a0, false, b0lo, (short)0, s0, false, false);
    s0 = __builtin_amdgcn_wmma_f32_16x16x32_bf16(false, a1, false, b0hi, (short)0, s0, false, false);
    s1 = __builtin_amdgcn_wmma_f32_16x16x32_bf16(false, a0, false, b1lo, (short)0, s1, false, false);
    s1 = __builtin_amdgcn_wmma_f32_16x16x32_bf16(false, a1, false, b1hi, (short)0, s1, false, false);

    // online softmax; lanes in half hf hold stats for rows v + 8*hf
#pragma unroll
    for (int v = 0; v < 8; ++v) {
      int qr_abs = qbase + v + 8 * hf;
      float x0 = s0[v] * scl; if (kb0 + lr > qr_abs)      x0 = -1e30f;
      float x1 = s1[v] * scl; if (kb0 + 16 + lr > qr_abs) x1 = -1e30f;
      float rmax = hmax16(fmaxf(x0, x1));
      float newm = fmaxf(mrow[v], rmax);
      float p0 = __expf(x0 - newm);
      float p1 = __expf(x1 - newm);
      float rescale = __expf(mrow[v] - newm);
      lrow[v] = lrow[v] * rescale + hsum16(p0 + p1);
      mrow[v] = newm;
      o0[v] *= rescale; o1[v] *= rescale; o2[v] *= rescale; o3[v] *= rescale;
      pl[(v + 8 * hf) * 32 + lr]      = (bf16)p0;
      pl[(v + 8 * hf) * 32 + 16 + lr] = (bf16)p1;
    }
    // reload P as A-fragment (row = lr, 32 key columns)
    v16bf pa = load_a_frag(pl + lr * 32, koff);
    // V B-fragments: elems e = V[key = kb0 + 16*hf + e][col], contiguous in vT
    const bf16* vrow = vp + kb0 + hf * 16;
    v16bf vb0 = *(const v16bf*)(vrow + (size_t)(lr)      * SEQ);
    v16bf vb1 = *(const v16bf*)(vrow + (size_t)(16 + lr) * SEQ);
    v16bf vb2 = *(const v16bf*)(vrow + (size_t)(32 + lr) * SEQ);
    v16bf vb3 = *(const v16bf*)(vrow + (size_t)(48 + lr) * SEQ);
    o0 = __builtin_amdgcn_wmma_f32_16x16x32_bf16(false, pa, false, vb0, (short)0, o0, false, false);
    o1 = __builtin_amdgcn_wmma_f32_16x16x32_bf16(false, pa, false, vb1, (short)0, o1, false, false);
    o2 = __builtin_amdgcn_wmma_f32_16x16x32_bf16(false, pa, false, vb2, (short)0, o2, false, false);
    o3 = __builtin_amdgcn_wmma_f32_16x16x32_bf16(false, pa, false, vb3, (short)0, o3, false, false);
  }

  // normalize + SiLU, write bf16 o_act (B, L, H)
  v8f oo[4] = {o0, o1, o2, o3};
#pragma unroll
  for (int c = 0; c < 4; ++c) {
#pragma unroll
    for (int v = 0; v < 8; ++v) {
      int l = qbase + v + 8 * hf;
      float val = oo[c][v] / lrow[v];
      float act = val / (1.f + __expf(-val));
      o_act[((size_t)(b * SEQ + l)) * HDIM + h * HD + c * 16 + lr] = (bf16)act;
    }
  }
}

// ---------------------------------------------------------------------------
// launch
// ---------------------------------------------------------------------------
extern "C" void kernel_launch(void* const* d_in, const int* in_sizes, int n_in,
                              void* d_out, int out_size, void* d_ws, size_t ws_size,
                              hipStream_t stream) {
  const float* x     = (const float*)d_in[0];
  const float* w_qkv = (const float*)d_in[1];
  const float* b_qkv = (const float*)d_in[2];
  const float* w_out = (const float*)d_in[3];
  const float* b_out = (const float*)d_in[4];
  float* out = (float*)d_out;

  char* ws = (char*)d_ws;
  size_t off = 0;
  auto alloc = [&](size_t bytes) -> void* {
    void* p = ws + off;
    off += (bytes + 255) & ~(size_t)255;
    return p;
  };

  const int ML = BATCH * SEQ;  // 4096
  bf16* xb    = (bf16*)alloc((size_t)ML * HDIM * 2);
  bf16* wqkvb = (bf16*)alloc((size_t)QKV_N * HDIM * 2);
  bf16* woutb = (bf16*)alloc((size_t)HDIM * HDIM * 2);
  float* qkv  = (float*)alloc((size_t)ML * QKV_N * 4);
  bf16* qb    = (bf16*)alloc((size_t)BATCH * NHEADS * SEQ * HD * 2);
  bf16* kb    = (bf16*)alloc((size_t)BATCH * NHEADS * SEQ * HD * 2);
  bf16* vTb   = (bf16*)alloc((size_t)BATCH * NHEADS * HD * SEQ * 2);
  bf16* oact  = (bf16*)alloc((size_t)ML * HDIM * 2);

  int nx = ML * HDIM, nwq = QKV_N * HDIM, nwo = HDIM * HDIM;
  cvt_bf16_kernel<<<(nx  + 255) / 256, 256, 0, stream>>>(x, xb, nx);
  cvt_bf16_kernel<<<(nwq + 255) / 256, 256, 0, stream>>>(w_qkv, wqkvb, nwq);
  cvt_bf16_kernel<<<(nwo + 255) / 256, 256, 0, stream>>>(w_out, woutb, nwo);

  // QKV GEMM: (4096 x 1024) @ (1024 x 3072): 32 x 24 = 768 block tiles
  gemm_lds_kernel<<<(ML / 128) * (QKV_N / 128), 256, 0, stream>>>(
      xb, wqkvb, b_qkv, qkv, ML, QKV_N, HDIM);

  // RoPE + layout transform
  {
    int n = BATCH * SEQ * NHEADS * 32;
    rope_kernel<<<(n + 255) / 256, 256, 0, stream>>>(qkv, qb, kb, vTb);
  }

  // Flash attention + SiLU
  attn_kernel<<<BATCH * NHEADS * (SEQ / 128), 256, 0, stream>>>(qb, kb, vTb, oact);

  // Output projection: (4096 x 1024) @ (1024 x 1024): 32 x 8 = 256 block tiles
  gemm_lds_kernel<<<(ML / 128) * (HDIM / 128), 256, 0, stream>>>(
      oact, woutb, b_out, out, ML, HDIM, HDIM);
}
